// Net_39032662786372
// MI455X (gfx1250) — matrix operand order repaired and checked by reading, not compiled, not run on hardware
//
#include <hip/hip_runtime.h>
#include <hip/hip_bf16.h>

typedef __attribute__((ext_vector_type(16))) __bf16 v16bf;
typedef __attribute__((ext_vector_type(8)))  float  v8f;

#define F_DIM 128
#define LDS_STRIDE 132   // 128 + 4 pad to break bank alignment

// ---------------------------------------------------------------------------
// Convert both 128x128 f32 weight matrices to bf16.
// ---------------------------------------------------------------------------
__global__ void cvt_weights_kernel(const float* __restrict__ w1,
                                   const float* __restrict__ w2,
                                   __bf16* __restrict__ w1b,
                                   __bf16* __restrict__ w2b) {
    int i = blockIdx.x * blockDim.x + threadIdx.x;
    if (i < F_DIM * F_DIM) {
        w1b[i] = (__bf16)w1[i];
        w2b[i] = (__bf16)w2[i];
    }
}

// ---------------------------------------------------------------------------
// Fused 2-layer GEMM:  out = (in @ w1^T + b1) @ w2^T + b2
// in/out: [n,128] f32.  w1b/w2b: [128,128] bf16 row-major ([out,in]).
// One wave per 16-row tile, 4 waves per block.
// WMMA bf16 16x16x32, f32 accumulate.
// ---------------------------------------------------------------------------
__global__ void gemm_fused_kernel(const float* __restrict__ in,
                                  const __bf16* __restrict__ w1b,
                                  const float* __restrict__ b1,
                                  const __bf16* __restrict__ w2b,
                                  const float* __restrict__ b2,
                                  float* __restrict__ out,
                                  int n) {
    __shared__ float lds[4][16 * LDS_STRIDE];

    const int wave = threadIdx.x >> 5;
    const int lane = threadIdx.x & 31;
    const int half = lane >> 4;       // 0: lanes 0-15, 1: lanes 16-31
    const int l16  = lane & 15;

    const int row0 = (blockIdx.x * 4 + wave) * 16;

    // ---- Layer 1: acc[nt] = A(in rows) x B(w1) ----
    v8f acc[8];
    const v8f vzero = {};
#pragma unroll
    for (int nt = 0; nt < 8; ++nt) acc[nt] = vzero;

    // clamped row for A loads (stores are guarded later; duplicate rows harmless)
    int arow = row0 + l16;
    if (arow > n - 1) arow = n - 1;
    const float* __restrict__ arp = in + (size_t)arow * F_DIM;

#pragma unroll
    for (int kk = 0; kk < 4; ++kk) {
        // A fragment: 16-bit A layout, lane half selects K sub-blocks
        const float* p = arp + kk * 32 + half * 16;
        v16bf a;
#pragma unroll
        for (int i = 0; i < 8; ++i) {
            a[i]     = (__bf16)p[i];        // K = kk*32 + half*16 + i
            a[8 + i] = (__bf16)p[8 + i];    // K = kk*32 + half*16 + 8 + i
        }
#pragma unroll
        for (int nt = 0; nt < 8; ++nt) {
            // B fragment: lane holds column (nt*16 + l16), K = kk*32 + half*16 .. +15
            const int col = nt * 16 + l16;
            v16bf b = *(const v16bf*)(w1b + col * F_DIM + kk * 32 + half * 16);
            acc[nt] = __builtin_amdgcn_wmma_f32_16x16x32_bf16(
                false, a, false, b, (short)0, acc[nt], false, false);
        }
    }

    // Stage h1 + b1 into LDS (C layout -> row-major)
    float* slab = &lds[wave][0];
#pragma unroll
    for (int nt = 0; nt < 8; ++nt) {
        const int col = nt * 16 + l16;
        const float bv = b1[col];
#pragma unroll
        for (int r = 0; r < 8; ++r) {
            const int m = r + 8 * half;   // C layout: lanes 0-15 -> M=r, 16-31 -> M=r+8
            slab[m * LDS_STRIDE + col] = acc[nt][r] + bv;
        }
    }
    __syncthreads();

    // ---- Layer 2: acc[nt] = A(h1 from LDS) x B(w2) ----
#pragma unroll
    for (int nt = 0; nt < 8; ++nt) acc[nt] = vzero;

    const float* __restrict__ a2p = slab + l16 * LDS_STRIDE;
#pragma unroll
    for (int kk = 0; kk < 4; ++kk) {
        const float* p = a2p + kk * 32 + half * 16;
        v16bf a;
#pragma unroll
        for (int i = 0; i < 8; ++i) {
            a[i]     = (__bf16)p[i];
            a[8 + i] = (__bf16)p[8 + i];
        }
#pragma unroll
        for (int nt = 0; nt < 8; ++nt) {
            const int col = nt * 16 + l16;
            v16bf b = *(const v16bf*)(w2b + col * F_DIM + kk * 32 + half * 16);
            acc[nt] = __builtin_amdgcn_wmma_f32_16x16x32_bf16(
                false, a, false, b, (short)0, acc[nt], false, false);
        }
    }

    // Store h2 + b2 (guard rows)
#pragma unroll
    for (int nt = 0; nt < 8; ++nt) {
        const int col = nt * 16 + l16;
        const float bv = b2[col];
#pragma unroll
        for (int r = 0; r < 8; ++r) {
            const int m = r + 8 * half;
            const int grow = row0 + m;
            if (grow < n) out[(size_t)grow * F_DIM + col] = acc[nt][r] + bv;
        }
    }
}

// ---------------------------------------------------------------------------
// out = src  (float4 copy; supplies the self-loop contribution)
// ---------------------------------------------------------------------------
__global__ void init_copy_kernel(const float4* __restrict__ src,
                                 float4* __restrict__ dst, int n4) {
    int i = blockIdx.x * blockDim.x + threadIdx.x;
    if (i < n4) dst[i] = src[i];
}

// ---------------------------------------------------------------------------
// Edge scatter-add: out[dst] += tmp[src].  32 threads per edge, 4 feats each.
// ---------------------------------------------------------------------------
__global__ void scatter_edges_kernel(const int* __restrict__ src,
                                     const int* __restrict__ dst,
                                     const float* __restrict__ tmp,
                                     float* __restrict__ out, int ne) {
    int idx = blockIdx.x * blockDim.x + threadIdx.x;
    int e = idx >> 5;
    if (e >= ne) return;
    int c = (idx & 31) << 2;
    int s = src[e];
    int d = dst[e];
    const float4 v = *(const float4*)(tmp + (size_t)s * F_DIM + c);
    float* o = out + (size_t)d * F_DIM + c;
    atomicAdd(o + 0, v.x);
    atomicAdd(o + 1, v.y);
    atomicAdd(o + 2, v.z);
    atomicAdd(o + 3, v.w);
}

// ---------------------------------------------------------------------------
// In-place row-wise log_softmax. One wave (32 lanes) per row, 4 feats/lane.
// ---------------------------------------------------------------------------
__global__ void log_softmax_kernel(float* __restrict__ x, int n) {
    const int wave = threadIdx.x >> 5;
    const int lane = threadIdx.x & 31;
    const int row = blockIdx.x * 4 + wave;
    if (row >= n) return;

    float4 v = *(const float4*)(x + (size_t)row * F_DIM + lane * 4);

    float m = fmaxf(fmaxf(v.x, v.y), fmaxf(v.z, v.w));
#pragma unroll
    for (int off = 16; off > 0; off >>= 1)
        m = fmaxf(m, __shfl_xor(m, off));

    float s = expf(v.x - m) + expf(v.y - m) + expf(v.z - m) + expf(v.w - m);
#pragma unroll
    for (int off = 16; off > 0; off >>= 1)
        s += __shfl_xor(s, off);

    const float shift = m + logf(s);
    v.x -= shift; v.y -= shift; v.z -= shift; v.w -= shift;
    *(float4*)(x + (size_t)row * F_DIM + lane * 4) = v;
}

// ---------------------------------------------------------------------------
// Launch
// ---------------------------------------------------------------------------
extern "C" void kernel_launch(void* const* d_in, const int* in_sizes, int n_in,
                              void* d_out, int out_size, void* d_ws, size_t ws_size,
                              hipStream_t stream) {
    const float* x  = (const float*)d_in[0];
    const int*   ei = (const int*)d_in[1];
    const float* w1 = (const float*)d_in[2];
    const float* b1 = (const float*)d_in[3];
    const float* w2 = (const float*)d_in[4];
    const float* b2 = (const float*)d_in[5];
    float* outp = (float*)d_out;

    const int n  = in_sizes[0] / F_DIM;     // 50000 nodes
    const int ne = in_sizes[1] / 2;         // 600000 edges
    const int* src = ei;
    const int* dst = ei + ne;

    const size_t nb = (size_t)n * F_DIM * sizeof(float);
    float*  t0  = (float*)d_ws;                              // GEMM output buffer
    float*  g0  = (float*)((char*)d_ws + nb);                // conv1 aggregation
    __bf16* w1b = (__bf16*)((char*)d_ws + 2 * nb);
    __bf16* w2b = w1b + F_DIM * F_DIM;

    const int n4 = n * F_DIM / 4;
    const int gemm_blocks    = (n + 63) / 64;
    const int copy_blocks    = (n4 + 255) / 256;
    const int scatter_blocks = (ne * 32 + 255) / 256;
    const int sm_blocks      = (n + 3) / 4;

    // weights -> bf16
    cvt_weights_kernel<<<(F_DIM * F_DIM + 255) / 256, 256, 0, stream>>>(w1, w2, w1b, w2b);

    // ---- conv 1 ----
    gemm_fused_kernel<<<gemm_blocks, 128, 0, stream>>>(x, w1b, b1, w2b, b2, t0, n);
    init_copy_kernel<<<copy_blocks, 256, 0, stream>>>((const float4*)t0, (float4*)g0, n4);
    scatter_edges_kernel<<<scatter_blocks, 256, 0, stream>>>(src, dst, t0, g0, ne);

    // ---- conv 2 ----
    gemm_fused_kernel<<<gemm_blocks, 128, 0, stream>>>(g0, w1b, b1, w2b, b2, t0, n);
    init_copy_kernel<<<copy_blocks, 256, 0, stream>>>((const float4*)t0, (float4*)outp, n4);
    scatter_edges_kernel<<<scatter_blocks, 256, 0, stream>>>(src, dst, t0, outp, ne);

    // ---- log_softmax in place on d_out ----
    log_softmax_kernel<<<sm_blocks, 128, 0, stream>>>(outp, n);
}
